// CustomConv2d_35356170781267
// MI455X (gfx1250) — compile-verified, matrix operand order checked by reading, not run on hardware
//
#include <hip/hip_runtime.h>

// ---- problem constants (from reference) ----
#define IN_C   128
#define OUT_C  256
#define KSZ    3
#define BATCH  16
#define HH     112
#define WW     112
#define KTOT   (IN_C * KSZ * KSZ)   // 1152
#define HWPROD (HH * WW)            // 12544
#define M_TOTAL (BATCH * HWPROD)    // 200704 spatial outputs

// ---- tiling ----
#define BM 128    // cout tile  (WMMA M axis)
#define BN 128    // spatial tile (WMMA N axis)
#define KC 32     // cin chunk per WMMA step
#define LDT 40    // padded LDS row pitch in ushorts (80 B, keeps b128 frags 16B-aligned)

typedef __attribute__((ext_vector_type(16))) __bf16 v16bf;
typedef __attribute__((ext_vector_type(8)))  float  v8f;
typedef __attribute__((ext_vector_type(2)))  __bf16 bf16x2;
typedef __attribute__((ext_vector_type(2)))  float  f32x2;

union FragBF { v16bf v; uint4 q[2]; };
union AccU   { v8f   v; float f[8]; };

// packed f32->bf16 (RNE) -> v_cvt_pk_bf16_f32
__device__ __forceinline__ unsigned int cvt2_bf16(float a, float b) {
    f32x2 f; f.x = a; f.y = b;
    bf16x2 h = __builtin_convertvector(f, bf16x2);
    union { bf16x2 h; unsigned int u; } c; c.h = h;
    return c.u;
}

__device__ __forceinline__ int clampi(int v, int lo, int hi) {
    return v < lo ? lo : (v > hi ? hi : v);   // v_med3_i32
}

__global__ __launch_bounds__(256)
void conv3x3_implicit_gemm_bf16wmma(const float* __restrict__ x,
                                    const float* __restrict__ w,
                                    const float* __restrict__ bias,
                                    float* __restrict__ out)
{
    // A tile: weights [cout 128][cin 32]   (bf16, padded rows)
    // B tile: im2col  [spatial 128][cin 32] (bf16, padded rows)
    __shared__ unsigned short sW[BM * LDT];   // 10240 B
    __shared__ unsigned short sX[BN * LDT];   // 10240 B

    const int tid  = threadIdx.x;
    const int lane = tid & 31;
    const int wave = tid >> 5;
    const int wm   = wave >> 1;   // cout quarter 0..3 (32 couts each)
    const int wn   = wave & 1;    // spatial half 0..1 (64 spatial each)

    const int tileS = blockIdx.x / (OUT_C / BM);   // 0..1567
    const int tileC = blockIdx.x % (OUT_C / BM);   // 0..1
    const int c0 = tileC * BM;
    const int s0 = tileS * BN;

    // ---- W staging mapping: thread -> (cout row, 16-cin group) ----
    const int wr   = tid >> 1;          // 0..127
    const int wkg  = (tid & 1) * 16;    // {0,16}
    const float* wbase = w + (size_t)(c0 + wr) * KTOT;   // + cin*9 + r later

    // ---- X staging mapping: thread -> (one spatial position, 16-cin group) ----
    const int xs   = tid >> 1;          // spatial row in tile, 0..127
    const int xkg  = (tid & 1) * 16;    // {0,16}
    const int sgx  = s0 + xs;
    const int xb   = sgx / HWPROD;
    const int xrem = sgx - xb * HWPROD;
    const int xoh  = xrem / WW;
    const int xow  = xrem - xoh * WW;
    const float* xbat = x + (size_t)xb * IN_C * HH * WW;

    v8f acc[2][4] = {{{}, {}, {}, {}}, {{}, {}, {}, {}}};

    // fragment lane addressing (ISA 16-bit A/B layout):
    // lanes 0-15: row = lane,    K in {0..7, 16..23}
    // lanes16-31: row = lane-16, K in {8..15, 24..31}
    const int frow = (lane & 15);
    const int koff = (lane >> 4) * 8;
    const unsigned short* fA0 = &sW[(wm * 32 + frow) * LDT + koff];
    const unsigned short* fB0 = &sX[(wn * 64 + frow) * LDT + koff];

    // ---- filter-position loop (no per-element k decode anywhere) ----
    #pragma unroll 1
    for (int r = 0; r < KSZ * KSZ; ++r) {
        const int kh = r / 3;
        const int kw = r - kh * 3;

        // one boundary predicate per thread per r; clamped coords keep loads in-bounds
        const int ih = xoh + kh - 1;
        const int iw = xow + kw - 1;
        const bool ok = ((unsigned)ih < (unsigned)HH) & ((unsigned)iw < (unsigned)WW);
        const int ihc = clampi(ih, 0, HH - 1);
        const int iwc = clampi(iw, 0, WW - 1);
        const float* xpos = xbat + (size_t)ihc * WW + iwc;  // + cin*HWPROD later
        const float* wpos = wbase + r;

        #pragma unroll 1
        for (int cin0 = 0; cin0 < IN_C; cin0 += KC) {
            // ---- stage weight tile: 16 stride-9 loads, pack, two b128 stores ----
            {
                const float* wp = wpos + (size_t)(cin0 + wkg) * (KSZ * KSZ);
                float a[16];
                #pragma unroll
                for (int j = 0; j < 16; ++j) a[j] = wp[j * (KSZ * KSZ)];
                uint4 p0, p1;
                p0.x = cvt2_bf16(a[0],  a[1]);
                p0.y = cvt2_bf16(a[2],  a[3]);
                p0.z = cvt2_bf16(a[4],  a[5]);
                p0.w = cvt2_bf16(a[6],  a[7]);
                p1.x = cvt2_bf16(a[8],  a[9]);
                p1.y = cvt2_bf16(a[10], a[11]);
                p1.z = cvt2_bf16(a[12], a[13]);
                p1.w = cvt2_bf16(a[14], a[15]);
                uint4* dst = reinterpret_cast<uint4*>(&sW[wr * LDT + wkg]);
                dst[0] = p0;
                dst[1] = p1;
            }
            // ---- stage im2col tile: 16 stride-HWPROD loads (imm offsets),
            //      branchless zero-select, two b128 stores ----
            {
                const float* bp = xpos + (size_t)(cin0 + xkg) * HWPROD;
                float v[16];
                #pragma unroll
                for (int j = 0; j < 16; ++j) {
                    float l = bp[(size_t)j * HWPROD];   // unconditional, clamped-safe
                    v[j] = ok ? l : 0.0f;               // v_cndmask, no exec branches
                }
                uint4 p0, p1;
                p0.x = cvt2_bf16(v[0],  v[1]);
                p0.y = cvt2_bf16(v[2],  v[3]);
                p0.z = cvt2_bf16(v[4],  v[5]);
                p0.w = cvt2_bf16(v[6],  v[7]);
                p1.x = cvt2_bf16(v[8],  v[9]);
                p1.y = cvt2_bf16(v[10], v[11]);
                p1.z = cvt2_bf16(v[12], v[13]);
                p1.w = cvt2_bf16(v[14], v[15]);
                uint4* dst = reinterpret_cast<uint4*>(&sX[xs * LDT + xkg]);
                dst[0] = p0;
                dst[1] = p1;
                // prefetch next cin chunk of the 102 MB x stream
                if (cin0 + KC < IN_C)
                    __builtin_prefetch(bp + (size_t)KC * HWPROD, 0, 1);
            }
            __syncthreads();

            // ---- fragments + 8 WMMAs (2 A frags x 4 B frags) ----
            FragBF fa[2];
            #pragma unroll
            for (int a = 0; a < 2; ++a) {
                const uint4* pa = reinterpret_cast<const uint4*>(fA0 + a * 16 * LDT);
                fa[a].q[0] = pa[0]; fa[a].q[1] = pa[2];
            }
            #pragma unroll
            for (int t = 0; t < 4; ++t) {
                FragBF fb;
                const uint4* pb = reinterpret_cast<const uint4*>(fB0 + t * 16 * LDT);
                fb.q[0] = pb[0]; fb.q[1] = pb[2];
                #pragma unroll
                for (int a = 0; a < 2; ++a)
                    acc[a][t] = __builtin_amdgcn_wmma_f32_16x16x32_bf16(
                                    false, fa[a].v, false, fb.v, (short)0,
                                    acc[a][t], false, false);
            }
            __syncthreads();
        }
    }

    // ---- epilogue: bias add + coalesced NCHW stores ----
    // C/D layout: N = lane&15 (spatial, consecutive ow across lanes),
    //             M = vgpr + 8*(lane>>4) (cout)
    const int clane8 = (lane >> 4) * 8;
    #pragma unroll
    for (int t = 0; t < 4; ++t) {
        const int s_local = wn * 64 + t * 16 + (lane & 15);
        const int sgl  = s0 + s_local;
        const int ob   = sgl / HWPROD;
        const int orem = sgl - ob * HWPROD;
        const int ooh  = orem / WW;
        const int oow  = orem - ooh * WW;
        float* obase = out + (((size_t)ob * OUT_C) * HH + ooh) * (size_t)WW + oow;
        #pragma unroll
        for (int a = 0; a < 2; ++a) {
            AccU acu; acu.v = acc[a][t];
            #pragma unroll
            for (int vv = 0; vv < 8; ++vv) {
                const int cout = c0 + wm * 32 + a * 16 + clane8 + vv;
                obase[(size_t)cout * HWPROD] = acu.f[vv] + bias[cout];
            }
        }
    }
}

extern "C" void kernel_launch(void* const* d_in, const int* in_sizes, int n_in,
                              void* d_out, int out_size, void* d_ws, size_t ws_size,
                              hipStream_t stream) {
    const float* x    = (const float*)d_in[0];
    const float* wgt  = (const float*)d_in[1];
    const float* bias = (const float*)d_in[2];
    float* outp       = (float*)d_out;

    // M_TOTAL/BN = 1568 spatial tiles, OUT_C/BM = 2 cout tiles
    dim3 grid((M_TOTAL / BN) * (OUT_C / BM));
    conv3x3_implicit_gemm_bf16wmma<<<grid, 256, 0, stream>>>(x, wgt, bias, outp);
}
